// MoE_SwiGLU_3659312136855
// MI455X (gfx1250) — matrix operand order repaired and checked by reading, not compiled, not run on hardware
//
#include <hip/hip_runtime.h>
#include <hip/hip_bf16.h>

// ---------------- problem constants ----------------
#define DIM      1024
#define NEXP     8
#define HID      2048
#define NTOK     (4 * 4096)         // B*T = 16384 tokens
#define OUTN     (NTOK * DIM + 1)   // out + aux_loss scalar
#define TILE_M   32
#define MAXTILES (NTOK / TILE_M)    // 512 tiles per expert worst case
#define WAVES    8                  // 256-thread blocks = 8 wave32

typedef __bf16 bf16;
typedef __bf16 bf16x16 __attribute__((ext_vector_type(16)));
typedef float  f32x8   __attribute__((ext_vector_type(8)));
typedef int    i32x4   __attribute__((ext_vector_type(4)));

// ---------------- WMMA helpers ----------------
__device__ __forceinline__ f32x8 wmma_bf16(bf16x16 a, bf16x16 b, f32x8 c) {
    // D = A(16x32 bf16) * B(32x16 bf16) + C(16x16 f32)
    return __builtin_amdgcn_wmma_f32_16x16x32_bf16(
        /*neg_a=*/false, a, /*neg_b=*/false, b,
        /*c_mod=*/(short)0, c, /*reuse_a=*/false, /*reuse_b=*/false);
}

// A-matrix fragment, row-major source.
// Lane l<16: row=l,   K = {kb..kb+7, kb+16..kb+23}
// Lane l>=16: row=l-16, K = {kb+8..kb+15, kb+24..kb+31}
// `p` must already point at row_base + (lane>>4)*8 + kb.
__device__ __forceinline__ bf16x16 load_fragA(const bf16* p) {
    union { bf16x16 v; i32x4 q[2]; } u;
    u.q[0] = *(const i32x4*)(p);
    u.q[1] = *(const i32x4*)(p + 16);
    return u.v;
}

// B-matrix fragment from weights stored [N][K] row-major.
// Lane l: column N = (l&15); K = {kb+(l>>4)*8..+7, kb+16+(l>>4)*8..+7}.
// `p` must already point at (nbase+(lane&15))*stride + (lane>>4)*8 + kb.
__device__ __forceinline__ bf16x16 load_fragB(const bf16* p) {
    union { bf16x16 v; i32x4 q[2]; } u;
    u.q[0] = *(const i32x4*)(p);
    u.q[1] = *(const i32x4*)(p + 16);
    return u.v;
}

// ---------------- kernel 0: init out + counters ----------------
__global__ void init_kernel(float* __restrict__ out, int n, int* __restrict__ cnt) {
    int i = blockIdx.x * blockDim.x + threadIdx.x;
    if (i < n) out[i] = 0.0f;
    if (i < NEXP) cnt[i] = 0;
}

// ---------------- kernel 1: fp32 -> bf16 transpose ----------------
// in: [NEXP][R][C] fp32  ->  out: [NEXP][C][R] bf16
__global__ void transpose_bf16_kernel(const float* __restrict__ in,
                                      bf16* __restrict__ out, int R, int C) {
    long long idx = (long long)blockIdx.x * blockDim.x + threadIdx.x;
    long long total = (long long)NEXP * R * C;
    if (idx >= total) return;
    int rc  = R * C;
    int e   = (int)(idx / rc);
    int rem = (int)(idx - (long long)e * rc);
    int c   = rem / R;            // out is r-fastest => coalesced writes
    int r   = rem - c * R;
    out[idx] = (bf16)in[(long long)e * rc + (long long)r * C + c];
}

// ---------------- kernel 2: router (one wave32 per token) ----------------
__global__ void router_kernel(const float* __restrict__ x,
                              const float* __restrict__ rw,
                              int* __restrict__ cnt,
                              int* __restrict__ btok,
                              float* __restrict__ bgate) {
    int lane = threadIdx.x & 31;
    int tok  = blockIdx.x * (blockDim.x >> 5) + (threadIdx.x >> 5);
    if (tok >= NTOK) return;

    float xv[32];
#pragma unroll
    for (int i = 0; i < 32; ++i)
        xv[i] = x[(size_t)tok * DIM + lane + 32 * i];

    float lg[NEXP];
#pragma unroll
    for (int e = 0; e < NEXP; ++e) {
        float a = 0.0f;
#pragma unroll
        for (int i = 0; i < 32; ++i)
            a += xv[i] * rw[e * DIM + lane + 32 * i];
#pragma unroll
        for (int off = 16; off > 0; off >>= 1)
            a += __shfl_down(a, off, 32);
        lg[e] = a;                 // valid on lane 0
    }

    if (lane == 0) {
        int i0 = 0; float b0 = lg[0];
#pragma unroll
        for (int e = 1; e < NEXP; ++e)
            if (lg[e] > b0) { b0 = lg[e]; i0 = e; }
        int i1 = -1; float b1 = -3.4e38f;
#pragma unroll
        for (int e = 0; e < NEXP; ++e)
            if (e != i0 && lg[e] > b1) { b1 = lg[e]; i1 = e; }
        // renormalized top-2 softmax weights (softmax denom cancels)
        float eb = __expf(b1 - b0);
        float g0 = 1.0f / (1.0f + eb);
        float g1 = eb / (1.0f + eb);

        int p0 = atomicAdd(&cnt[i0], 1);
        btok[i0 * NTOK + p0]  = tok;
        bgate[i0 * NTOK + p0] = g0;
        int p1 = atomicAdd(&cnt[i1], 1);
        btok[i1 * NTOK + p1]  = tok;
        bgate[i1 * NTOK + p1] = g1;
    }
}

// ---------------- kernel 3: fused expert FFN ----------------
// One block = one (expert, 32-token tile). LDS: x tile (bf16, 64KB) +
// h tile (bf16, 128KB) + gates/tokens. Fits in the 320KB WGP LDS.
// Each wave owns n-tiles and computes BOTH 16-row m-tiles per k-step so
// every B fragment (global/L2 traffic) feeds 2 WMMAs.
#define SMEM_BYTES (TILE_M * DIM * 2 + TILE_M * HID * 2 + TILE_M * 4 + TILE_M * 4)

__global__ void ffn_kernel(const float* __restrict__ x,
                           const bf16* __restrict__ w1t,   // [e][h][d]
                           const bf16* __restrict__ w2t,   // [e][h][d]
                           const bf16* __restrict__ w3t,   // [e][d][h]
                           const int* __restrict__ cnt,
                           const int* __restrict__ btok,
                           const float* __restrict__ bgate,
                           float* __restrict__ out) {
    extern __shared__ char smem[];
    bf16*  xs  = (bf16*)smem;                                   // [32][1024]
    bf16*  hs  = (bf16*)(smem + TILE_M * DIM * 2);              // [32][2048]
    float* gsh = (float*)(smem + TILE_M * DIM * 2 + TILE_M * HID * 2);
    int*   tsh = (int*)(gsh + TILE_M);

    int e  = blockIdx.x / MAXTILES;
    int mt = blockIdx.x % MAXTILES;
    int count = cnt[e];
    if (mt * TILE_M >= count) return;   // uniform: whole block exits

    int tid = threadIdx.x;
    if (tid < TILE_M) {
        int gi = mt * TILE_M + tid;
        int tk = (gi < count) ? btok[e * NTOK + gi] : -1;
        tsh[tid] = (tk < 0) ? 0 : tk;
        gsh[tid] = (gi < count) ? bgate[e * NTOK + gi] : 0.0f;
    }
    __syncthreads();

    // Stage gathered x rows into LDS as bf16 (coalesced per row).
    for (int idx = tid; idx < TILE_M * DIM; idx += blockDim.x) {
        int r = idx >> 10, c = idx & (DIM - 1);
        xs[idx] = (bf16)x[(size_t)tsh[r] * DIM + c];
    }
    __syncthreads();

    int wave = tid >> 5, lane = tid & 31;
    int lcol = lane & 15, half = lane >> 4;

    // ---- Phase 1: h = silu(x@W1) * (x@W2), h -> LDS (bf16) ----
    const bf16* W1 = w1t + (size_t)e * HID * DIM;
    const bf16* W2 = w2t + (size_t)e * HID * DIM;
    for (int nt = wave; nt < HID / 16; nt += WAVES) {   // 16 n-tiles per wave
        f32x8 a1m0 = {0.f,0.f,0.f,0.f,0.f,0.f,0.f,0.f};
        f32x8 a2m0 = {0.f,0.f,0.f,0.f,0.f,0.f,0.f,0.f};
        f32x8 a1m1 = {0.f,0.f,0.f,0.f,0.f,0.f,0.f,0.f};
        f32x8 a2m1 = {0.f,0.f,0.f,0.f,0.f,0.f,0.f,0.f};
        const bf16* ap0 = xs + (size_t)lcol * DIM + half * 8;          // m-tile 0
        const bf16* ap1 = ap0 + (size_t)16 * DIM;                      // m-tile 1
        const bf16* b1p = W1 + (size_t)(nt * 16 + lcol) * DIM + half * 8;
        const bf16* b2p = W2 + (size_t)(nt * 16 + lcol) * DIM + half * 8;
        for (int k = 0; k < DIM; k += 32) {
            bf16x16 b1 = load_fragB(b1p + k);       // each B frag feeds 2 WMMAs
            bf16x16 b2 = load_fragB(b2p + k);
            bf16x16 a0 = load_fragA(ap0 + k);
            bf16x16 a1 = load_fragA(ap1 + k);
            a1m0 = wmma_bf16(a0, b1, a1m0);
            a2m0 = wmma_bf16(a0, b2, a2m0);
            a1m1 = wmma_bf16(a1, b1, a1m1);
            a2m1 = wmma_bf16(a1, b2, a2m1);
        }
        int col = nt * 16 + lcol;
#pragma unroll
        for (int j = 0; j < 8; ++j) {
            int r0 = half * 8 + j;                 // m-tile 0 row
            float a = a1m0[j], b = a2m0[j];
            float s = a / (1.0f + __expf(-a));     // silu
            hs[r0 * HID + col] = (bf16)(s * b);
            int r1 = 16 + half * 8 + j;            // m-tile 1 row
            float a2 = a1m1[j], b2v = a2m1[j];
            float s2 = a2 / (1.0f + __expf(-a2));
            hs[r1 * HID + col] = (bf16)(s2 * b2v);
        }
    }
    __syncthreads();

    // ---- Phase 2: out += gate * (h @ W3) via fp32 atomics ----
    const bf16* W3 = w3t + (size_t)e * DIM * HID;
    for (int nt = wave; nt < DIM / 16; nt += WAVES) {   // 8 n-tiles per wave
        f32x8 accm0 = {0.f,0.f,0.f,0.f,0.f,0.f,0.f,0.f};
        f32x8 accm1 = {0.f,0.f,0.f,0.f,0.f,0.f,0.f,0.f};
        const bf16* ap0 = hs + (size_t)lcol * HID + half * 8;
        const bf16* ap1 = ap0 + (size_t)16 * HID;
        const bf16* bp  = W3 + (size_t)(nt * 16 + lcol) * HID + half * 8;
        for (int k = 0; k < HID; k += 32) {
            bf16x16 bfr = load_fragB(bp + k);
            bf16x16 a0  = load_fragA(ap0 + k);
            bf16x16 a1  = load_fragA(ap1 + k);
            accm0 = wmma_bf16(a0, bfr, accm0);
            accm1 = wmma_bf16(a1, bfr, accm1);
        }
        int col = nt * 16 + lcol;
#pragma unroll
        for (int j = 0; j < 8; ++j) {
            int r0 = half * 8 + j;
            unsafeAtomicAdd(&out[(size_t)tsh[r0] * DIM + col], gsh[r0] * accm0[j]);
            int r1 = 16 + half * 8 + j;
            unsafeAtomicAdd(&out[(size_t)tsh[r1] * DIM + col], gsh[r1] * accm1[j]);
        }
    }
}

// ---------------- host launcher ----------------
extern "C" void kernel_launch(void* const* d_in, const int* in_sizes, int n_in,
                              void* d_out, int out_size, void* d_ws, size_t ws_size,
                              hipStream_t stream) {
    (void)in_sizes; (void)n_in; (void)out_size;
    const float* x  = (const float*)d_in[0];
    const float* rw = (const float*)d_in[1];
    const float* w1 = (const float*)d_in[2];
    const float* w2 = (const float*)d_in[3];
    const float* w3 = (const float*)d_in[4];
    float* out = (float*)d_out;

    // workspace layout (~102 MB; harness-provided scratch)
    char* ws = (char*)d_ws; (void)ws_size;
    size_t wbytes = (size_t)NEXP * DIM * HID * sizeof(bf16);   // 32 MB each
    bf16*  w1t   = (bf16*)(ws);
    bf16*  w2t   = (bf16*)(ws + wbytes);
    bf16*  w3t   = (bf16*)(ws + 2 * wbytes);
    int*   btok  = (int*)  (ws + 3 * wbytes);
    float* bgate = (float*)(ws + 3 * wbytes + (size_t)NEXP * NTOK * 4);
    int*   cnt   = (int*)  (ws + 3 * wbytes + 2 * (size_t)NEXP * NTOK * 4);

    // 0) zero outputs + counters
    init_kernel<<<(OUTN + 255) / 256, 256, 0, stream>>>(out, OUTN, cnt);

    // 1) weight transpose + bf16 convert (amortized over all tokens)
    int wtotal = NEXP * DIM * HID;                              // 16.8M
    transpose_bf16_kernel<<<wtotal / 256, 256, 0, stream>>>(w1, w1t, DIM, HID);
    transpose_bf16_kernel<<<wtotal / 256, 256, 0, stream>>>(w2, w2t, DIM, HID);
    transpose_bf16_kernel<<<wtotal / 256, 256, 0, stream>>>(w3, w3t, HID, DIM);

    // 2) router + top-2 scatter (one wave32 per token)
    router_kernel<<<NTOK / WAVES, 256, 0, stream>>>(x, rw, cnt, btok, bgate);

    // 3) fused expert FFN: 8 experts x 512 token-tiles (inactive tiles exit)
    ffn_kernel<<<NEXP * MAXTILES, 256, SMEM_BYTES, stream>>>(
        x, w1t, w2t, w3t, cnt, btok, bgate, out);
}
